// PointNetFeaturePropagation_82841329205664
// MI455X (gfx1250) — compile-verified, matrix operand order checked by reading, not run on hardware
//
#include <hip/hip_runtime.h>
#include <hip/hip_bf16.h>
#include <cstdint>
#include <cstddef>

// Problem dims (fixed by setup_inputs)
#define B_    8
#define N_    8192
#define S_    2048
#define D1_   128
#define D2_   256
#define CIN_  384
#define C1_   256
#define C2_   256
#define ROWS_ (B_ * N_)   // 65536

typedef __attribute__((ext_vector_type(16))) __bf16 v16bf;
typedef __attribute__((ext_vector_type(8)))  float  v8f;

// ---------------------------------------------------------------------------
// K0: float -> bf16 convert (weights)
// ---------------------------------------------------------------------------
__global__ void cvt_bf16_kernel(const float* __restrict__ in, __bf16* __restrict__ out, int n) {
    int i = blockIdx.x * 256 + threadIdx.x;
    if (i < n) out[i] = (__bf16)in[i];
}

// ---------------------------------------------------------------------------
// K1: tiled transpose (B,C,S) -> (B,S,C), fp32 (for points2 -> f2)
// ---------------------------------------------------------------------------
__global__ void transpose_cs_kernel(const float* __restrict__ in, float* __restrict__ out,
                                    int C, int S) {
    __shared__ float tile[32][33];
    int b  = blockIdx.z;
    int c0 = blockIdx.x * 32;
    int s0 = blockIdx.y * 32;
    int tx = threadIdx.x, ty = threadIdx.y;  // 32 x 8
    #pragma unroll
    for (int i = ty; i < 32; i += 8)
        tile[i][tx] = in[((size_t)b * C + (c0 + i)) * S + (s0 + tx)];
    __syncthreads();
    #pragma unroll
    for (int i = ty; i < 32; i += 8)
        out[((size_t)b * S + (s0 + i)) * C + (c0 + tx)] = tile[tx][i];
}

// ---------------------------------------------------------------------------
// K2: 3-NN search. xyz2 for the batch staged in LDS (3*2048*4 = 24 KB).
// Produces idx (3 per row) and normalized inverse-distance weights.
// ---------------------------------------------------------------------------
__global__ void knn3_kernel(const float* __restrict__ xyz1, const float* __restrict__ xyz2,
                            int* __restrict__ idx, float* __restrict__ wgt) {
    __shared__ float sx[S_], sy[S_], sz[S_];
    int b = blockIdx.y;
    int n = blockIdx.x * 256 + threadIdx.x;
    for (int s = threadIdx.x; s < S_; s += 256) {
        sx[s] = xyz2[((size_t)b * 3 + 0) * S_ + s];
        sy[s] = xyz2[((size_t)b * 3 + 1) * S_ + s];
        sz[s] = xyz2[((size_t)b * 3 + 2) * S_ + s];
    }
    __syncthreads();
    float px = xyz1[((size_t)b * 3 + 0) * N_ + n];
    float py = xyz1[((size_t)b * 3 + 1) * N_ + n];
    float pz = xyz1[((size_t)b * 3 + 2) * N_ + n];
    float d0 = 3.4e38f, d1 = 3.4e38f, d2 = 3.4e38f;
    int   i0 = 0, i1 = 0, i2 = 0;
    #pragma unroll 4
    for (int s = 0; s < S_; ++s) {
        float dx = px - sx[s], dy = py - sy[s], dz = pz - sz[s];
        float d = dx * dx + dy * dy + dz * dz;
        if (d < d2) {
            if (d < d1) {
                if (d < d0) { d2 = d1; i2 = i1; d1 = d0; i1 = i0; d0 = d; i0 = s; }
                else        { d2 = d1; i2 = i1; d1 = d;  i1 = s; }
            } else          { d2 = d;  i2 = s; }
        }
    }
    float r0 = 1.0f / (d0 + 1e-8f);
    float r1 = 1.0f / (d1 + 1e-8f);
    float r2 = 1.0f / (d2 + 1e-8f);
    float rs = 1.0f / (r0 + r1 + r2);
    size_t row = (size_t)b * N_ + n;
    idx[row * 3 + 0] = i0; idx[row * 3 + 1] = i1; idx[row * 3 + 2] = i2;
    wgt[row * 3 + 0] = r0 * rs; wgt[row * 3 + 1] = r1 * rs; wgt[row * 3 + 2] = r2 * rs;
}

// ---------------------------------------------------------------------------
// K3: tiled transpose points1 (B,128,N) into columns [0,128) of Xbf (bf16)
// ---------------------------------------------------------------------------
__global__ void p1_to_x_kernel(const float* __restrict__ p1, __bf16* __restrict__ X) {
    __shared__ float tile[32][33];
    int b  = blockIdx.z;
    int c0 = blockIdx.x * 32;
    int n0 = blockIdx.y * 32;
    int tx = threadIdx.x, ty = threadIdx.y;  // 32 x 8
    #pragma unroll
    for (int i = ty; i < 32; i += 8)
        tile[i][tx] = p1[((size_t)b * D1_ + (c0 + i)) * N_ + (n0 + tx)];
    __syncthreads();
    #pragma unroll
    for (int i = ty; i < 32; i += 8)
        X[((size_t)b * N_ + (n0 + i)) * CIN_ + (c0 + tx)] = (__bf16)tile[tx][i];
}

// ---------------------------------------------------------------------------
// K4: weighted 3-NN interpolation into columns [128,384) of Xbf
// one block per output row; 256 threads = feature channels (coalesced rows of f2)
// ---------------------------------------------------------------------------
__global__ void interp_kernel(const float* __restrict__ f2, const int* __restrict__ idx,
                              const float* __restrict__ wgt, __bf16* __restrict__ X) {
    size_t row = blockIdx.x;
    int c = threadIdx.x;
    int b = (int)(row >> 13);   // row / N_
    int i0 = idx[row * 3 + 0], i1 = idx[row * 3 + 1], i2 = idx[row * 3 + 2];
    float w0 = wgt[row * 3 + 0], w1 = wgt[row * 3 + 1], w2 = wgt[row * 3 + 2];
    const float* base = f2 + (size_t)b * S_ * D2_;
    float v = w0 * base[(size_t)i0 * D2_ + c]
            + w1 * base[(size_t)i1 * D2_ + c]
            + w2 * base[(size_t)i2 * D2_ + c];
    X[row * CIN_ + D1_ + c] = (__bf16)v;
}

// ---------------------------------------------------------------------------
// K5: bf16 WMMA GEMM.  C[rows x ncols] = A[rows x K](bf16) * W[ncols x K]^T + bias
// Each wave computes a 16x32 output tile (two 16x16 accumulators sharing one
// A-tile load): 3 x b128 loads per WMMA instead of 4.
// Block = 8 waves arranged 4 (row-tiles) x 2 (col-pairs) = 64 rows x 64 cols.
// A/B VGPR layout per CDNA5 ISA: lane%16 = M (resp. N) index, lane/16 = K-half;
// vector elements 0..7 -> K (e%8)+8*kh, elements 8..15 -> 16+(e%8)+8*kh.
// ---------------------------------------------------------------------------
__global__ void gemm_bf16_wmma_kernel(const __bf16* __restrict__ A, const __bf16* __restrict__ W,
                                      const float* __restrict__ bias, float* __restrict__ C,
                                      int K, int ncols) {
    const int lane = threadIdx.x & 31;
    const int wave = threadIdx.x >> 5;
    const int m0 = blockIdx.x * 64 + (wave & 3) * 16;
    const int n0 = blockIdx.y * 64 + (wave >> 2) * 32;   // wave covers cols [n0, n0+32)
    const int mi = lane & 15;   // M index for A, N index for B
    const int kh = lane >> 4;   // K half selector

    const __bf16* pa  = A + (size_t)(m0 + mi)      * K + 8 * kh;
    const __bf16* pb0 = W + (size_t)(n0 + mi)      * K + 8 * kh;
    const __bf16* pb1 = W + (size_t)(n0 + 16 + mi) * K + 8 * kh;

    v8f acc0 = {};
    v8f acc1 = {};
    for (int k0 = 0; k0 < K; k0 += 32) {
        union { v16bf v; uint4 q[2]; } a, b0, b1;
        a.q[0]  = *(const uint4*)(pa);
        a.q[1]  = *(const uint4*)(pa + 16);
        b0.q[0] = *(const uint4*)(pb0);
        b0.q[1] = *(const uint4*)(pb0 + 16);
        b1.q[0] = *(const uint4*)(pb1);
        b1.q[1] = *(const uint4*)(pb1 + 16);
        pa += 32; pb0 += 32; pb1 += 32;
        __builtin_prefetch(pa + 32, 0, 3);   // gfx1250 global_prefetch_b8, WGP scope
        acc0 = __builtin_amdgcn_wmma_f32_16x16x32_bf16(
            false, a.v, false, b0.v, (short)0, acc0, false, false);
        acc1 = __builtin_amdgcn_wmma_f32_16x16x32_bf16(
            false, a.v, false, b1.v, (short)0, acc1, false, false);
    }

    const int col0 = n0 + mi;
    const int col1 = n0 + 16 + mi;
    const float bv0 = bias[col0];
    const float bv1 = bias[col1];
    float* pc = C + (size_t)(m0 + 8 * kh) * ncols;
    #pragma unroll
    for (int r = 0; r < 8; ++r) {
        pc[(size_t)r * ncols + col0] = acc0[r] + bv0;
        pc[(size_t)r * ncols + col1] = acc1[r] + bv1;
    }
}

// ---------------------------------------------------------------------------
// K6: zero small buffer
// ---------------------------------------------------------------------------
__global__ void zero_kernel(float* __restrict__ p, int n) {
    int i = blockIdx.x * 256 + threadIdx.x;
    if (i < n) p[i] = 0.0f;
}

// ---------------------------------------------------------------------------
// K7: per-channel sum / sum-of-squares (256 channels), block-partial + atomic
// ---------------------------------------------------------------------------
__global__ void stats_kernel(const float* __restrict__ Y, float* __restrict__ sums, int rowsPerBlock) {
    int c  = threadIdx.x;                    // 256 channels
    size_t r0 = (size_t)blockIdx.x * rowsPerBlock;
    float s = 0.0f, ss = 0.0f;
    for (int r = 0; r < rowsPerBlock; ++r) {
        float v = Y[(r0 + r) * 256 + c];
        s += v; ss += v * v;
    }
    atomicAdd(&sums[c], s);
    atomicAdd(&sums[256 + c], ss);
}

// ---------------------------------------------------------------------------
// K8: fold BN into per-channel scale/shift: y = x*sA[c] + sB[c]
// ---------------------------------------------------------------------------
__global__ void finalize_stats_kernel(const float* __restrict__ sums,
                                      const float* __restrict__ gamma,
                                      const float* __restrict__ beta,
                                      float* __restrict__ sAB, float inv_count) {
    int c = threadIdx.x;    // 256
    float mean = sums[c] * inv_count;
    float var  = sums[256 + c] * inv_count - mean * mean;
    float sA   = gamma[c] * rsqrtf(var + 1e-5f);
    sAB[c]       = sA;
    sAB[256 + c] = beta[c] - mean * sA;
}

// ---------------------------------------------------------------------------
// K9: apply BN+ReLU and convert to bf16 (A-matrix for the second GEMM)
// ---------------------------------------------------------------------------
__global__ void bn_relu_bf16_kernel(const float* __restrict__ Y, const float* __restrict__ sAB,
                                    __bf16* __restrict__ out) {
    size_t i = (size_t)blockIdx.x * 256 + threadIdx.x;
    int c = (int)(i & 255);
    float v = Y[i] * sAB[c] + sAB[256 + c];
    out[i] = (__bf16)fmaxf(v, 0.0f);
}

// ---------------------------------------------------------------------------
// K10: apply BN+ReLU and write transposed output (B,256,N), tiled via LDS
// ---------------------------------------------------------------------------
__global__ void bn_relu_out_kernel(const float* __restrict__ Y, const float* __restrict__ sAB,
                                   float* __restrict__ out) {
    __shared__ float tile[32][33];
    int b  = blockIdx.z;
    int o0 = blockIdx.x * 32;
    int n0 = blockIdx.y * 32;
    int tx = threadIdx.x, ty = threadIdx.y;  // 32 x 8
    #pragma unroll
    for (int i = ty; i < 32; i += 8) {
        int o = o0 + tx;
        float v = Y[((size_t)b * N_ + (n0 + i)) * 256 + o];
        tile[i][tx] = fmaxf(v * sAB[o] + sAB[256 + o], 0.0f);
    }
    __syncthreads();
    #pragma unroll
    for (int i = ty; i < 32; i += 8)
        out[((size_t)b * 256 + (o0 + i)) * N_ + (n0 + tx)] = tile[tx][i];
}

// ---------------------------------------------------------------------------
// Host-side launch
// ---------------------------------------------------------------------------
extern "C" void kernel_launch(void* const* d_in, const int* in_sizes, int n_in,
                              void* d_out, int out_size, void* d_ws, size_t ws_size,
                              hipStream_t stream) {
    const float* xyz1    = (const float*)d_in[0];
    const float* xyz2    = (const float*)d_in[1];
    const float* points1 = (const float*)d_in[2];
    const float* points2 = (const float*)d_in[3];
    const float* W1      = (const float*)d_in[4];
    const float* b1      = (const float*)d_in[5];
    const float* g1      = (const float*)d_in[6];
    const float* beta1   = (const float*)d_in[7];
    const float* W2      = (const float*)d_in[8];
    const float* b2      = (const float*)d_in[9];
    const float* g2      = (const float*)d_in[10];
    const float* beta2   = (const float*)d_in[11];
    float* out = (float*)d_out;

    // ---- workspace layout (256-byte aligned slabs) ----
    char* ws = (char*)d_ws;
    size_t off = 0;
    auto alloc = [&](size_t bytes) { char* p = ws + off; off += (bytes + 255) & ~(size_t)255; return p; };
    float*  f2     = (float*)alloc((size_t)B_ * S_ * D2_ * 4);          // 16.8 MB
    int*    idx    = (int*)  alloc((size_t)ROWS_ * 3 * 4);              // 0.8 MB
    float*  wgt    = (float*)alloc((size_t)ROWS_ * 3 * 4);              // 0.8 MB
    __bf16* Xbf    = (__bf16*)alloc((size_t)ROWS_ * CIN_ * 2);          // 50.3 MB
    float*  Y1     = (float*)alloc((size_t)ROWS_ * C1_ * 4);            // 67.1 MB (reused as Y2)
    __bf16* A2bf   = (__bf16*)alloc((size_t)ROWS_ * C1_ * 2);           // 33.6 MB
    __bf16* W1bf   = (__bf16*)alloc((size_t)C1_ * CIN_ * 2);
    __bf16* W2bf   = (__bf16*)alloc((size_t)C2_ * C1_ * 2);
    float*  stats1 = (float*)alloc(512 * 4);
    float*  sAB1   = (float*)alloc(512 * 4);
    float*  stats2 = (float*)alloc(512 * 4);
    float*  sAB2   = (float*)alloc(512 * 4);
    float*  Y2     = Y1;   // GEMM2 overwrites the dead Y1 buffer

    // ---- weights to bf16 ----
    cvt_bf16_kernel<<<(C1_ * CIN_ + 255) / 256, 256, 0, stream>>>(W1, W1bf, C1_ * CIN_);
    cvt_bf16_kernel<<<(C2_ * C1_  + 255) / 256, 256, 0, stream>>>(W2, W2bf, C2_ * C1_);

    // ---- points2 -> f2 (B,S,256) ----
    transpose_cs_kernel<<<dim3(D2_ / 32, S_ / 32, B_), dim3(32, 8), 0, stream>>>(points2, f2, D2_, S_);

    // ---- 3-NN + weights ----
    knn3_kernel<<<dim3(N_ / 256, B_), 256, 0, stream>>>(xyz1, xyz2, idx, wgt);

    // ---- build X = [points1^T | interp] as bf16 ----
    p1_to_x_kernel<<<dim3(D1_ / 32, N_ / 32, B_), dim3(32, 8), 0, stream>>>(points1, Xbf);
    interp_kernel<<<ROWS_, 256, 0, stream>>>(f2, idx, wgt, Xbf);

    // ---- layer 1: GEMM (bf16 WMMA, fp32 accum) + bias ----
    gemm_bf16_wmma_kernel<<<dim3(ROWS_ / 64, C1_ / 64), 256, 0, stream>>>(Xbf, W1bf, b1, Y1, CIN_, C1_);

    // ---- BN1 stats + fused scale/shift+ReLU -> bf16 A for layer 2 ----
    zero_kernel<<<2, 256, 0, stream>>>(stats1, 512);
    stats_kernel<<<ROWS_ / 256, 256, 0, stream>>>(Y1, stats1, 256);
    finalize_stats_kernel<<<1, 256, 0, stream>>>(stats1, g1, beta1, sAB1, 1.0f / (float)ROWS_);
    bn_relu_bf16_kernel<<<(unsigned)((size_t)ROWS_ * C1_ / 256), 256, 0, stream>>>(Y1, sAB1, A2bf);

    // ---- layer 2: GEMM + bias (writes over Y1 slab) ----
    gemm_bf16_wmma_kernel<<<dim3(ROWS_ / 64, C2_ / 64), 256, 0, stream>>>(A2bf, W2bf, b2, Y2, C1_, C2_);

    // ---- BN2 stats + fused BN+ReLU into transposed output ----
    zero_kernel<<<2, 256, 0, stream>>>(stats2, 512);
    stats_kernel<<<ROWS_ / 256, 256, 0, stream>>>(Y2, stats2, 256);
    finalize_stats_kernel<<<1, 256, 0, stream>>>(stats2, g2, beta2, sAB2, 1.0f / (float)ROWS_);
    bn_relu_out_kernel<<<dim3(C2_ / 32, N_ / 32, B_), dim3(32, 8), 0, stream>>>(Y2, sAB2, out);

    (void)in_sizes; (void)n_in; (void)out_size; (void)ws_size;
}
